// AttnBlockpp_30777735643620
// MI455X (gfx1250) — compile-verified, matrix operand order checked by reading, not run on hardware
//
#include <hip/hip_runtime.h>
#include <math.h>

typedef _Float16 v16h __attribute__((ext_vector_type(16)));
typedef _Float16 v8h  __attribute__((ext_vector_type(8)));
typedef float    v8f  __attribute__((ext_vector_type(8)));
typedef unsigned int u32x4 __attribute__((ext_vector_type(4)));
typedef int          i32x4 __attribute__((ext_vector_type(4)));
typedef int          i32x8 __attribute__((ext_vector_type(8)));

#define WMMA(a, b, c) __builtin_amdgcn_wmma_f32_16x16x32_f16(false, (a), false, (b), (short)0, (c), false, false)

#define B_N   8
#define HW_N  4096          // H*W = 64*64
#define C_N   256
#define M_TOT (B_N * HW_N)  // 32768 rows total
#define GRPS  32
#define CG    8             // channels per group
#define SM_SCALE 0.0625f    // 256^-0.5
#define INV_SQRT2 0.70710678118654752f

// ---- Tensor Data Mover availability (probe-hedged across toolchains) ----
#if defined(__has_builtin)
#if __has_builtin(__builtin_amdgcn_tensor_load_to_lds) && __has_builtin(__builtin_amdgcn_s_wait_tensorcnt)
#define USE_TDM 1
#endif
#endif
#ifndef USE_TDM
#define USE_TDM 0
#endif

// Load a 16(row) x 32(K) f16 fragment per ISA 7.12.2 layout.
// lane L: row = L&15, hi = L>>4; elems 0..7 = [row][hi*8 + 0..7],
// elems 8..15 = [row][16 + hi*8 + 0..7]. K must be contiguous in memory.
static __device__ __forceinline__ v16h frag16x32(const _Float16* p, int ld, int lane) {
    const int r  = lane & 15;
    const int hi = (lane >> 4) & 1;
    const _Float16* base = p + (size_t)r * ld + hi * 8;
    v8h lo = *(const v8h*)(base);
    v8h hh = *(const v8h*)(base + 16);
    v16h out;
#pragma unroll
    for (int i = 0; i < 8; ++i) { out[i] = lo[i]; out[i + 8] = hh[i]; }
    return out;
}

#if USE_TDM
// Issue a 2D TDM tile load: tensor rows of tensor_d0 contiguous elems (2B each),
// row stride = stride0 elems; tile = tile_d0 x tile_d1 starting at gptr -> LDS.
// D# layout per CDNA5 ISA section 8 (group0: count/lds/global/type; group1: dims).
static __device__ __forceinline__ void tdm_load_tile_2d(unsigned lds_addr,
        const _Float16* gptr, unsigned tensor_d0, unsigned tensor_d1,
        unsigned tile_d0, unsigned tile_d1, unsigned stride0) {
    const unsigned long long ga = (unsigned long long)gptr;
    u32x4 g0;
    g0[0] = 1u;                                            // count=1, user desc
    g0[1] = lds_addr;                                      // LDS byte address
    g0[2] = (unsigned)(ga & 0xFFFFFFFFu);                  // global_addr[31:0]
    g0[3] = (unsigned)((ga >> 32) & 0x1FFFFFFu) | (2u << 30); // global_addr[56:32] | type=2
    i32x8 g1;
    g1[0] = 0x00010000;                                    // wg_mask=0, data_size=1 (2 bytes)
    g1[1] = (int)((tensor_d0 & 0xFFFFu) << 16);            // tensor_dim0[15:0]
    g1[2] = (int)(((tensor_d0 >> 16) & 0xFFFFu) | ((tensor_d1 & 0xFFFFu) << 16));
    g1[3] = (int)(((tensor_d1 >> 16) & 0xFFFFu) | ((tile_d0 & 0xFFFFu) << 16));
    g1[4] = (int)(tile_d1 & 0xFFFFu);                      // tile_dim1, tile_dim2=0
    g1[5] = (int)stride0;                                  // tensor_dim0_stride[31:0]
    g1[6] = 0;                                             // stride0 hi, stride1 lo
    g1[7] = 0;
    i32x4 z4 = {0, 0, 0, 0};
#if __clang_major__ >= 23
    i32x8 z8 = {0, 0, 0, 0, 0, 0, 0, 0};
    __builtin_amdgcn_tensor_load_to_lds(g0, g1, z4, z4, z8, 0);
#else
    __builtin_amdgcn_tensor_load_to_lds(g0, g1, z4, z4, 0);
#endif
}
#endif

// Stage one 32-key tile of k ([32][256] f16) and vT ([256][32] f16) into LDS.
static __device__ __forceinline__ void stage_tiles(_Float16* ldsk, _Float16* ldsv,
                                                   const _Float16* kb, const _Float16* vb,
                                                   int j0, int wave, int tid) {
#if USE_TDM
    if (wave == 0) {
        tdm_load_tile_2d((unsigned)(unsigned long long)ldsk, kb + (size_t)j0 * C_N,
                         C_N, HW_N, C_N, 32, C_N);
        tdm_load_tile_2d((unsigned)(unsigned long long)ldsv, vb + j0,
                         HW_N, C_N, 32, C_N, HW_N);
    }
#else
    for (int t = tid; t < (32 * C_N) / 8; t += 256)
        ((v8h*)ldsk)[t] = *(const v8h*)(kb + (size_t)j0 * C_N + t * 8);
    for (int t = tid; t < (C_N * 32) / 8; t += 256) {
        const int c = t >> 2, part = t & 3;
        ((v8h*)ldsv)[t] = *(const v8h*)(vb + (size_t)c * HW_N + j0 + part * 8);
    }
#endif
}

// ---------------- GroupNorm -> f16 ----------------
__global__ void gn_kernel(const float* __restrict__ x,
                          const float* __restrict__ scale,
                          const float* __restrict__ bias,
                          _Float16* __restrict__ h16) {
    const int b = blockIdx.x >> 5;
    const int g = blockIdx.x & 31;
    const float* xb = x + (size_t)b * HW_N * C_N + g * CG;
    _Float16*   hb = h16 + (size_t)b * HW_N * C_N + g * CG;

    float s = 0.f, ss = 0.f;
    for (int i = threadIdx.x; i < HW_N * CG; i += 256) {
        const int hw = i >> 3, c = i & 7;
        const float v = xb[(size_t)hw * C_N + c];
        s += v; ss += v * v;
    }
#pragma unroll
    for (int off = 16; off; off >>= 1) {
        s  += __shfl_xor(s,  off, 32);
        ss += __shfl_xor(ss, off, 32);
    }
    __shared__ float rs[8], rss[8];
    if ((threadIdx.x & 31) == 0) { rs[threadIdx.x >> 5] = s; rss[threadIdx.x >> 5] = ss; }
    __syncthreads();
    float S = 0.f, SS = 0.f;
#pragma unroll
    for (int i = 0; i < 8; ++i) { S += rs[i]; SS += rss[i]; }
    const float inv_n = 1.0f / (HW_N * CG);
    const float mean  = S * inv_n;
    const float var   = SS * inv_n - mean * mean;
    const float rstd  = rsqrtf(var + 1e-6f);

    for (int i = threadIdx.x; i < HW_N * CG; i += 256) {
        const int hw = i >> 3, c = i & 7;
        const float v = xb[(size_t)hw * C_N + c];
        hb[(size_t)hw * C_N + c] =
            (_Float16)((v - mean) * rstd * scale[g * CG + c] + bias[g * CG + c]);
    }
}

// ---------------- Weight transpose f32[c][d] -> f16[d][c] ----------------
__global__ void wt_kernel(const float* __restrict__ w, _Float16* __restrict__ wt) {
    const int d = blockIdx.x, c = threadIdx.x;
    wt[(size_t)d * C_N + c] = (_Float16)w[(size_t)c * C_N + d];
}

// ---------------- 1x1 projection GEMM: out[m][d] = A[m][:] . Wt[d][:] + bias[d] ----------------
__global__ void nin_kernel(const _Float16* __restrict__ A,
                           const _Float16* __restrict__ Wt,   // [d][c] f16
                           const float* __restrict__ bias,
                           _Float16* __restrict__ out,
                           int transpose_out) {
    const int wave = threadIdx.x >> 5, lane = threadIdx.x & 31;
    const int m0 = blockIdx.x * 16;
    const int d0 = blockIdx.y * 128 + wave * 16;

    const _Float16* arow = A  + (size_t)m0 * C_N;
    const _Float16* brow = Wt + (size_t)d0 * C_N;
    v8f acc = {};
#pragma unroll
    for (int k0 = 0; k0 < C_N; k0 += 32) {
        v16h af = frag16x32(arow + k0, C_N, lane);
        v16h bf = frag16x32(brow + k0, C_N, lane);
        acc = WMMA(af, bf, acc);
    }
    const int n = lane & 15, hi = lane >> 4;
    const float bv = bias[d0 + n];
#pragma unroll
    for (int v = 0; v < 8; ++v) {
        const int m = m0 + hi * 8 + v;
        const float val = acc[v] + bv;
        if (!transpose_out) {
            out[(size_t)m * C_N + d0 + n] = (_Float16)val;
        } else {
            const int b = m >> 12, nn = m & (HW_N - 1);
            out[((size_t)(b * C_N + d0 + n)) * HW_N + nn] = (_Float16)val;  // [b][c][n]
        }
    }
}

// ---------------- Flash attention with TDM double-buffered k/v tiles ----------------
__global__ void attn_kernel(const _Float16* __restrict__ q,   // [b][n][c]
                            const _Float16* __restrict__ k,   // [b][n][c]
                            const _Float16* __restrict__ vT,  // [b][c][n]
                            _Float16* __restrict__ o16) {     // [b][n][c]
    __shared__ __align__(16) _Float16 lds_k[2][32][C_N];   // 2 x 16KB key tiles
    __shared__ __align__(16) _Float16 lds_v[2][C_N][32];   // 2 x 16KB value tiles
    __shared__ __align__(16) _Float16 lds_p[8][16][32];    // per-wave prob staging

    const int wave = threadIdx.x >> 5, lane = threadIdx.x & 31;
    const int b = blockIdx.x >> 5;                  // 32 blocks per batch
    const int q0 = (blockIdx.x & 31) * 128 + wave * 16;

    const _Float16* qb = q  + ((size_t)b * HW_N + q0) * C_N;
    const _Float16* kb = k  + (size_t)b * HW_N * C_N;
    const _Float16* vb = vT + (size_t)b * C_N * HW_N;

    // Preload q fragments (16 rows x full C) into registers
    v16h qf[8];
#pragma unroll
    for (int i = 0; i < 8; ++i) qf[i] = frag16x32(qb + i * 32, C_N, lane);

    v8f o[16];
#pragma unroll
    for (int i = 0; i < 16; ++i) o[i] = (v8f){};
    float mrow[8], lrow[8];
#pragma unroll
    for (int v = 0; v < 8; ++v) { mrow[v] = -3.0e38f; lrow[v] = 0.f; }

    const int hi = lane >> 4, col = lane & 15;

    // Prologue: fill both buffers (TENSORcnt = 4 outstanding in TDM path)
    stage_tiles(&lds_k[0][0][0], &lds_v[0][0][0], kb, vb, 0,  wave, threadIdx.x);
    stage_tiles(&lds_k[1][0][0], &lds_v[1][0][0], kb, vb, 32, wave, threadIdx.x);

    for (int it = 0; it < HW_N / 32; ++it) {
        const int j_next = ((it + 2) * 32) & (HW_N - 1);   // always keep 2 in flight
        const int cur = it & 1;
#if USE_TDM
        if (wave == 0) __builtin_amdgcn_s_wait_tensorcnt(2);  // current buffer landed
#endif
        __syncthreads();

        // s = q . k^T for this 32-key tile (k from LDS)
        v8f s0 = {}, s1 = {};
#pragma unroll
        for (int i = 0; i < 8; ++i) {
            v16h k0f = frag16x32(&lds_k[cur][0][0]  + i * 32, C_N, lane);
            v16h k1f = frag16x32(&lds_k[cur][16][0] + i * 32, C_N, lane);
            s0 = WMMA(qf[i], k0f, s0);
            s1 = WMMA(qf[i], k1f, s1);
        }
        // online softmax update (row m = hi*8+v lives in one 16-lane half at acc idx v)
        float alpha[8];
#pragma unroll
        for (int v = 0; v < 8; ++v) {
            const float a = s0[v] * SM_SCALE;
            const float c = s1[v] * SM_SCALE;
            float r = fmaxf(a, c);
#pragma unroll
            for (int off = 1; off < 16; off <<= 1) r = fmaxf(r, __shfl_xor(r, off, 32));
            const float mn = fmaxf(mrow[v], r);
            alpha[v] = __expf(mrow[v] - mn);
            mrow[v] = mn;
            const float p0 = __expf(a - mn);
            const float p1 = __expf(c - mn);
            float rsum = p0 + p1;
#pragma unroll
            for (int off = 1; off < 16; off <<= 1) rsum += __shfl_xor(rsum, off, 32);
            lrow[v] = lrow[v] * alpha[v] + rsum;
            const int row = hi * 8 + v;
            lds_p[wave][row][col]      = (_Float16)p0;
            lds_p[wave][row][col + 16] = (_Float16)p1;
        }
        // rescale running accumulator
#pragma unroll
        for (int cc = 0; cc < 16; ++cc)
#pragma unroll
            for (int v = 0; v < 8; ++v) o[cc][v] *= alpha[v];

        asm volatile("s_wait_dscnt 0" ::: "memory");
        v16h pf = frag16x32(&lds_p[wave][0][0], 32, lane);

        // o += p . v  (K = 32 keys; B column = channel, keys contiguous in LDS v tile)
#pragma unroll
        for (int cc = 0; cc < 16; ++cc) {
            v16h vf = frag16x32(&lds_v[cur][cc * 16][0], 32, lane);
            o[cc] = WMMA(pf, vf, o[cc]);
        }

        __syncthreads();  // everyone done reading buffer `cur`
        stage_tiles(&lds_k[cur][0][0], &lds_v[cur][0][0], kb, vb, j_next, wave, threadIdx.x);
    }

    // normalize and store
    float linv[8];
#pragma unroll
    for (int v = 0; v < 8; ++v) linv[v] = 1.0f / lrow[v];
    _Float16* ob = o16 + ((size_t)b * HW_N + q0) * C_N;
#pragma unroll
    for (int cc = 0; cc < 16; ++cc)
#pragma unroll
        for (int v = 0; v < 8; ++v) {
            const int m = hi * 8 + v;
            ob[(size_t)m * C_N + cc * 16 + col] = (_Float16)(o[cc][v] * linv[v]);
        }
}

// ---------------- Final projection + residual: out = (x + A.Wf + bf) / sqrt(2) ----------------
__global__ void final_kernel(const _Float16* __restrict__ A,
                             const _Float16* __restrict__ Wt,
                             const float* __restrict__ bias,
                             const float* __restrict__ x,
                             float* __restrict__ out) {
    const int wave = threadIdx.x >> 5, lane = threadIdx.x & 31;
    const int m0 = blockIdx.x * 16;
    const int d0 = blockIdx.y * 128 + wave * 16;

    const _Float16* arow = A  + (size_t)m0 * C_N;
    const _Float16* brow = Wt + (size_t)d0 * C_N;
    v8f acc = {};
#pragma unroll
    for (int k0 = 0; k0 < C_N; k0 += 32) {
        v16h af = frag16x32(arow + k0, C_N, lane);
        v16h bf = frag16x32(brow + k0, C_N, lane);
        acc = WMMA(af, bf, acc);
    }
    const int n = lane & 15, hi = lane >> 4;
    const float bv = bias[d0 + n];
#pragma unroll
    for (int v = 0; v < 8; ++v) {
        const int m = m0 + hi * 8 + v;
        const size_t idx = (size_t)m * C_N + d0 + n;
        out[idx] = (x[idx] + acc[v] + bv) * INV_SQRT2;
    }
}

extern "C" void kernel_launch(void* const* d_in, const int* in_sizes, int n_in,
                              void* d_out, int out_size, void* d_ws, size_t ws_size,
                              hipStream_t stream) {
    const float* x  = (const float*)d_in[0];
    const float* gs = (const float*)d_in[1];
    const float* gb = (const float*)d_in[2];
    const float* Wq = (const float*)d_in[3];
    const float* bq = (const float*)d_in[4];
    const float* Wk = (const float*)d_in[5];
    const float* bk = (const float*)d_in[6];
    const float* Wv = (const float*)d_in[7];
    const float* bv = (const float*)d_in[8];
    const float* Wf = (const float*)d_in[9];
    const float* bf = (const float*)d_in[10];
    float* out = (float*)d_out;

    char* ws = (char*)d_ws;
    const size_t tensz = (size_t)M_TOT * C_N * sizeof(_Float16);  // 16 MB each
    _Float16* h16  = (_Float16*)(ws);
    _Float16* q16  = (_Float16*)(ws + 1 * tensz);
    _Float16* k16  = (_Float16*)(ws + 2 * tensz);
    _Float16* vT16 = (_Float16*)(ws + 3 * tensz);
    _Float16* a16  = (_Float16*)(ws + 4 * tensz);
    _Float16* Wq16 = (_Float16*)(ws + 5 * tensz);
    _Float16* Wk16 = Wq16 + C_N * C_N;
    _Float16* Wv16 = Wk16 + C_N * C_N;
    _Float16* Wf16 = Wv16 + C_N * C_N;

    gn_kernel<<<B_N * GRPS, 256, 0, stream>>>(x, gs, gb, h16);

    wt_kernel<<<C_N, C_N, 0, stream>>>(Wq, Wq16);
    wt_kernel<<<C_N, C_N, 0, stream>>>(Wk, Wk16);
    wt_kernel<<<C_N, C_N, 0, stream>>>(Wv, Wv16);
    wt_kernel<<<C_N, C_N, 0, stream>>>(Wf, Wf16);

    dim3 gemm_grid(M_TOT / 16, 2);
    nin_kernel<<<gemm_grid, 256, 0, stream>>>(h16, Wq16, bq, q16, 0);
    nin_kernel<<<gemm_grid, 256, 0, stream>>>(h16, Wk16, bk, k16, 0);
    nin_kernel<<<gemm_grid, 256, 0, stream>>>(h16, Wv16, bv, vT16, 1);  // store v transposed

    attn_kernel<<<M_TOT / 128, 256, 0, stream>>>(q16, k16, vT16, a16);

    final_kernel<<<gemm_grid, 256, 0, stream>>>(a16, Wf16, bf, x, out);
}
